// CustomModel_40114994545336
// MI455X (gfx1250) — matrix operand order repaired and checked by reading, not compiled, not run on hardware
//
#include <hip/hip_runtime.h>
#include <math.h>

// ---------------- model dims ----------------
#define SEQ   4096
#define HID   768
#define NHEAD 12
#define DHEAD 64
#define FFN   3072
#define NLAY  4
#define NCLS  2
#define WIN   128
#define NBLK  (SEQ / WIN)

typedef __attribute__((ext_vector_type(16))) _Float16 v16h;
typedef __attribute__((ext_vector_type(8)))  float    v8f;

union FragH {
    uint4     q[2];   // two 16B chunks (8 fp16 each)
    v16h      v;
    _Float16  h[16];
};

static __device__ inline v8f wmma_f16(v16h a, v16h b, v8f c) {
    // D = A(16x32 f16) * B(32x16 f16) + C(16x16 f32)
    return __builtin_amdgcn_wmma_f32_16x16x32_f16(
        /*neg_a=*/false, a, /*neg_b=*/false, b,
        /*c_mod=*/(short)0, c, /*reuse_a=*/false, /*reuse_b=*/false);
}

static __device__ inline v8f zero8() {
    v8f z = {0.f, 0.f, 0.f, 0.f, 0.f, 0.f, 0.f, 0.f};
    return z;
}

// -------- async global->LDS copies (gfx1250), with safe fallback --------
#if defined(__HIP_DEVICE_COMPILE__) && defined(__has_builtin)
#if __has_builtin(__builtin_amdgcn_global_load_async_to_lds_b128) && \
    __has_builtin(__builtin_amdgcn_s_wait_asynccnt)
#define HAVE_ASYNC_LDS 1
#endif
#endif
#ifndef HAVE_ASYNC_LDS
#define HAVE_ASYNC_LDS 0
#endif

// builtin signature (from compiler diagnostic): param0 = AS1 v4i*, 4 args total
typedef int v4i __attribute__((vector_size(4 * sizeof(int))));

// copy 16 bytes (8 fp16) global -> LDS
static __device__ inline void cp16(_Float16* lds_dst, const _Float16* gsrc) {
#if HAVE_ASYNC_LDS
    __builtin_amdgcn_global_load_async_to_lds_b128(
        (__attribute__((address_space(1))) v4i*)gsrc,
        (__attribute__((address_space(3))) v4i*)lds_dst, 0, 0);
#else
    *(uint4*)lds_dst = *(const uint4*)gsrc;
#endif
}

static __device__ inline void async_join() {
#if HAVE_ASYNC_LDS
    __builtin_amdgcn_s_wait_asynccnt(0);
#endif
}

// ---------------- weight fp32[K][N] -> fp16 transposed [N][K] ----------------
__global__ __launch_bounds__(256) void cvt_transpose_kernel(
    const float* __restrict__ src, _Float16* __restrict__ dst, int K, int N)
{
    const size_t idx = (size_t)blockIdx.x * 256 + threadIdx.x;
    if (idx >= (size_t)K * N) return;
    const int n = (int)(idx % N);
    const int k = (int)(idx / N);
    dst[(size_t)n * K + k] = (_Float16)src[idx];
}

// ---------------- embedding + layernorm (fp32 + fp16 copies) ----------------
__global__ __launch_bounds__(256) void embed_ln_kernel(
    const int* __restrict__ ids, const float* __restrict__ we,
    const float* __restrict__ pe, const float* __restrict__ g,
    const float* __restrict__ b, float* __restrict__ x,
    _Float16* __restrict__ x16)
{
    const int s = blockIdx.x, t = threadIdx.x;
    __shared__ float red[256];
    __shared__ float s_mean, s_rstd;
    const int id = ids[s];
    float v[HID / 256];
    float lsum = 0.f;
#pragma unroll
    for (int i = 0; i < HID / 256; ++i) {
        int h = t + i * 256;
        v[i] = we[(size_t)id * HID + h] + pe[(size_t)s * HID + h];
        lsum += v[i];
    }
    red[t] = lsum; __syncthreads();
    for (int o = 128; o > 0; o >>= 1) { if (t < o) red[t] += red[t + o]; __syncthreads(); }
    if (t == 0) s_mean = red[0] / HID;
    __syncthreads();
    const float m = s_mean;
    float lvar = 0.f;
#pragma unroll
    for (int i = 0; i < HID / 256; ++i) { float d = v[i] - m; lvar += d * d; }
    red[t] = lvar; __syncthreads();
    for (int o = 128; o > 0; o >>= 1) { if (t < o) red[t] += red[t + o]; __syncthreads(); }
    if (t == 0) s_rstd = rsqrtf(red[0] / HID + 1e-5f);
    __syncthreads();
    const float rs = s_rstd;
#pragma unroll
    for (int i = 0; i < HID / 256; ++i) {
        int h = t + i * 256;
        float y = (v[i] - m) * rs * g[h] + b[h];
        x[(size_t)s * HID + h] = y;
        x16[(size_t)s * HID + h] = (_Float16)y;
    }
}

// ---------------- residual add + layernorm (out may alias a) ----------------
__global__ __launch_bounds__(256) void add_ln_kernel(
    const float* __restrict__ a, const float* __restrict__ c,
    const float* __restrict__ g, const float* __restrict__ b,
    float* __restrict__ out, _Float16* __restrict__ out16)
{
    const int s = blockIdx.x, t = threadIdx.x;
    __shared__ float red[256];
    __shared__ float s_mean, s_rstd;
    float v[HID / 256];
    float lsum = 0.f;
#pragma unroll
    for (int i = 0; i < HID / 256; ++i) {
        int h = t + i * 256;
        v[i] = a[(size_t)s * HID + h] + c[(size_t)s * HID + h];
        lsum += v[i];
    }
    red[t] = lsum; __syncthreads();
    for (int o = 128; o > 0; o >>= 1) { if (t < o) red[t] += red[t + o]; __syncthreads(); }
    if (t == 0) s_mean = red[0] / HID;
    __syncthreads();
    const float m = s_mean;
    float lvar = 0.f;
#pragma unroll
    for (int i = 0; i < HID / 256; ++i) { float d = v[i] - m; lvar += d * d; }
    red[t] = lvar; __syncthreads();
    for (int o = 128; o > 0; o >>= 1) { if (t < o) red[t] += red[t + o]; __syncthreads(); }
    if (t == 0) s_rstd = rsqrtf(red[0] / HID + 1e-5f);
    __syncthreads();
    const float rs = s_rstd;
#pragma unroll
    for (int i = 0; i < HID / 256; ++i) {
        int h = t + i * 256;
        float y = (v[i] - m) * rs * g[h] + b[h];
        out[(size_t)s * HID + h] = y;
        out16[(size_t)s * HID + h] = (_Float16)y;
    }
}

// ---------------- WMMA GEMM: Y = act(A16[M,K] * B16t[N,K]^T + bias) ----------
// EPI:  0 = bias, 1 = bias + exact GELU
// OUTM: 0 = fp32 [M][N], 1 = fp16 [M][N], 2 = fp16 transposed [N][M]
template <int EPI, int OUTM>
__global__ __launch_bounds__(256) void gemm16_kernel(
    const _Float16* __restrict__ A16, const _Float16* __restrict__ B16t,
    const float* __restrict__ bias, void* __restrict__ Yout,
    int M, int K, int N)
{
    constexpr int BM = 128, BN = 128, BK = 32;
    __shared__ __align__(16) _Float16 As[2][BM][BK];   // 2 x 8KB
    __shared__ __align__(16) _Float16 Bs[2][BN][BK];   // 2 x 8KB

    const int t = threadIdx.x;
    const int wave = t >> 5, lane = t & 31, lrow = lane & 15, h2 = lane >> 4;
    const int wr = wave >> 1, wc = wave & 1;           // 4 x 2 wave grid
    const int m0 = blockIdx.y * BM, n0 = blockIdx.x * BN;

    // staging: 2 threads per row, 16 elements (2 x 16B) each, A and B
    const int srow = t >> 1, scol = (t & 1) * 16;
    const _Float16* asrc = A16 + (size_t)(m0 + srow) * K + scol;
    const _Float16* bsrc = B16t + (size_t)(n0 + srow) * K + scol;

    auto stage = [&](int buf, int k0) {
        cp16(&As[buf][srow][scol],     asrc + k0);
        cp16(&As[buf][srow][scol + 8], asrc + k0 + 8);
        cp16(&Bs[buf][srow][scol],     bsrc + k0);
        cp16(&Bs[buf][srow][scol + 8], bsrc + k0 + 8);
    };

    v8f acc[2][4];
#pragma unroll
    for (int i = 0; i < 2; ++i)
#pragma unroll
        for (int j = 0; j < 4; ++j) acc[i][j] = zero8();

    const int nk = K / BK;
    stage(0, 0);
    async_join();
    __syncthreads();

    for (int kt = 0; kt < nk; ++kt) {
        const int cur = kt & 1;
        if (kt + 1 < nk) stage(cur ^ 1, (kt + 1) * BK);   // overlap with compute

        FragH a[2], b[4];
#pragma unroll
        for (int mi = 0; mi < 2; ++mi) {
            const int mm = wr * 32 + mi * 16 + lrow;
            a[mi].q[0] = *(const uint4*)&As[cur][mm][8 * h2];      // K = 8h..8h+7
            a[mi].q[1] = *(const uint4*)&As[cur][mm][16 + 8 * h2]; // K = 16+8h..
        }
#pragma unroll
        for (int ni = 0; ni < 4; ++ni) {
            const int nn = wc * 64 + ni * 16 + lrow;
            b[ni].q[0] = *(const uint4*)&Bs[cur][nn][8 * h2];      // K = 8h..8h+7
            b[ni].q[1] = *(const uint4*)&Bs[cur][nn][16 + 8 * h2];
        }
#pragma unroll
        for (int mi = 0; mi < 2; ++mi)
#pragma unroll
            for (int ni = 0; ni < 4; ++ni)
                acc[mi][ni] = wmma_f16(a[mi].v, b[ni].v, acc[mi][ni]);

        async_join();
        __syncthreads();
    }

    // epilogue: C layout -> VGPR r holds row r + 8*h2, col = lane%16
#pragma unroll
    for (int mi = 0; mi < 2; ++mi)
#pragma unroll
        for (int ni = 0; ni < 4; ++ni) {
            const int gn = n0 + wc * 64 + ni * 16 + lrow;
            const float bv = bias[gn];
#pragma unroll
            for (int r = 0; r < 8; ++r) {
                const int gm = m0 + wr * 32 + mi * 16 + r + 8 * h2;
                float val = acc[mi][ni][r] + bv;
                if (EPI == 1) val = 0.5f * val * (1.f + erff(val * 0.70710678f));
                if (OUTM == 0)      ((float*)Yout)[(size_t)gm * N + gn] = val;
                else if (OUTM == 1) ((_Float16*)Yout)[(size_t)gm * N + gn] = (_Float16)val;
                else                ((_Float16*)Yout)[(size_t)gn * M + gm] = (_Float16)val;
            }
        }
}

// ---------------- banded flash attention (W = 128 each side) ----------------
// Q16,K16: fp16 [S][H]; Vt16: fp16 [H][S] (pre-transposed by V-proj GEMM)
__global__ __launch_bounds__(256) void attn_kernel(
    const _Float16* __restrict__ Q16, const _Float16* __restrict__ K16,
    const _Float16* __restrict__ Vt16, const int* __restrict__ amask,
    _Float16* __restrict__ O16)
{
    __shared__ __align__(16) _Float16 Ks[WIN][DHEAD];    // [key][d]   16KB
    __shared__ __align__(16) _Float16 Vt[DHEAD][WIN];    // [d][key]   16KB
    __shared__ __align__(16) _Float16 Ps[8][16][WIN];    // per-wave probs 32KB

    const int head = blockIdx.x, n = blockIdx.y;
    const int t = threadIdx.x;
    const int wave = t >> 5, lane = t & 31, lrow = lane & 15, h2 = lane >> 4;
    const int hc = head * DHEAD;
    const int qg = n * WIN + wave * 16;   // first query row of this wave's strip

    // Q fragments kept in registers for all 3 key chunks (K = 64 -> 2 ksteps)
    FragH qa[2];
    {
        const _Float16* qsrc = Q16 + (size_t)(qg + lrow) * HID + hc;
#pragma unroll
        for (int ks = 0; ks < 2; ++ks) {
            qa[ks].q[0] = *(const uint4*)(qsrc + ks * 32 + 8 * h2);
            qa[ks].q[1] = *(const uint4*)(qsrc + ks * 32 + 16 + 8 * h2);
        }
    }

    v8f o[4];
#pragma unroll
    for (int nt = 0; nt < 4; ++nt) o[nt] = zero8();
    float mr[8], lr[8];
#pragma unroll
    for (int r = 0; r < 8; ++r) { mr[r] = -1e30f; lr[r] = 0.f; }

    const int kkey = t >> 1, kcb = (t & 1) * 32;   // K staging: 2 thr / key row
    const int vd = t >> 2,  vcb = (t & 3) * 32;    // V staging: 4 thr / d row

    for (int c = 0; c < 3; ++c) {
        const int kb = (n - 1 + c) * WIN;          // chunk's first global key
        if (kb < 0 || kb >= SEQ) continue;         // block-uniform skip
        __syncthreads();                           // prev chunk done with K/V
        { // stage K chunk [128][64] and V^T chunk [64][128], 16B async copies
            const _Float16* ksrc = K16 + (size_t)(kb + kkey) * HID + hc + kcb;
#pragma unroll
            for (int j = 0; j < 32; j += 8) cp16(&Ks[kkey][kcb + j], ksrc + j);
            const _Float16* vsrc = Vt16 + (size_t)(hc + vd) * SEQ + kb + vcb;
#pragma unroll
            for (int j = 0; j < 32; j += 8) cp16(&Vt[vd][vcb + j], vsrc + j);
        }
        async_join();
        __syncthreads();

        // scores: 16 x 128 per wave = 8 tiles, 2 ksteps each
        v8f sc[8];
#pragma unroll
        for (int tt = 0; tt < 8; ++tt) {
            sc[tt] = zero8();
#pragma unroll
            for (int ks = 0; ks < 2; ++ks) {
                FragH b;
                b.q[0] = *(const uint4*)&Ks[tt * 16 + lrow][ks * 32 + 8 * h2];
                b.q[1] = *(const uint4*)&Ks[tt * 16 + lrow][ks * 32 + 16 + 8 * h2];
                sc[tt] = wmma_f16(qa[ks].v, b.v, sc[tt]);
            }
        }

        // scale + band / key-padding mask, chunk row max
        float cmax[8];
#pragma unroll
        for (int r = 0; r < 8; ++r) cmax[r] = -1e30f;
#pragma unroll
        for (int tt = 0; tt < 8; ++tt) {
            const int kk = kb + tt * 16 + lrow;
            const float km = (amask[kk] != 0) ? 0.f : -1e9f;
#pragma unroll
            for (int r = 0; r < 8; ++r) {
                const int qrow = qg + r + 8 * h2;
                const int rel = kk - qrow;
                float v2 = ((rel >= -WIN) && (rel <= WIN)) ? (sc[tt][r] * 0.125f + km) : -1e9f;
                sc[tt][r] = v2;
                cmax[r] = fmaxf(cmax[r], v2);
            }
        }
#pragma unroll
        for (int r = 0; r < 8; ++r)
#pragma unroll
            for (int off = 1; off < 16; off <<= 1)
                cmax[r] = fmaxf(cmax[r], __shfl_xor(cmax[r], off, 32));

        // online softmax update
        float alpha[8], rowsum[8];
#pragma unroll
        for (int r = 0; r < 8; ++r) {
            const float mnew = fmaxf(mr[r], cmax[r]);
            alpha[r] = __expf(mr[r] - mnew);
            mr[r] = mnew;
            rowsum[r] = 0.f;
        }
#pragma unroll
        for (int tt = 0; tt < 8; ++tt)
#pragma unroll
            for (int r = 0; r < 8; ++r) {
                const float p = __expf(sc[tt][r] - mr[r]);
                rowsum[r] += p;
                Ps[wave][r + 8 * h2][tt * 16 + lrow] = (_Float16)p;
            }
#pragma unroll
        for (int r = 0; r < 8; ++r) {
#pragma unroll
            for (int off = 1; off < 16; off <<= 1)
                rowsum[r] += __shfl_xor(rowsum[r], off, 32);
            lr[r] = lr[r] * alpha[r] + rowsum[r];
        }
#pragma unroll
        for (int nt = 0; nt < 4; ++nt)
#pragma unroll
            for (int r = 0; r < 8; ++r) o[nt][r] *= alpha[r];

        __syncthreads();   // probs visible in LDS

        // O += P (16x128) * V (128x64): 4 ksteps x 4 n-tiles
#pragma unroll
        for (int ks2 = 0; ks2 < 4; ++ks2) {
            FragH pa;
            pa.q[0] = *(const uint4*)&Ps[wave][lrow][ks2 * 32 + 8 * h2];
            pa.q[1] = *(const uint4*)&Ps[wave][lrow][ks2 * 32 + 16 + 8 * h2];
#pragma unroll
            for (int nt = 0; nt < 4; ++nt) {
                FragH vb;
                vb.q[0] = *(const uint4*)&Vt[nt * 16 + lrow][ks2 * 32 + 8 * h2];
                vb.q[1] = *(const uint4*)&Vt[nt * 16 + lrow][ks2 * 32 + 16 + 8 * h2];
                o[nt] = wmma_f16(pa.v, vb.v, o[nt]);
            }
        }
    }

    // write normalized fp16 output
#pragma unroll
    for (int nt = 0; nt < 4; ++nt)
#pragma unroll
        for (int r = 0; r < 8; ++r) {
            const int qrow = qg + r + 8 * h2;
            O16[(size_t)qrow * HID + hc + nt * 16 + lrow] = (_Float16)(o[nt][r] / lr[r]);
        }
}

// ---------------- head: x+pos, pooling score ----------------
__global__ __launch_bounds__(256) void head_pos_score_kernel(
    const float* __restrict__ x, const float* __restrict__ hpe,
    const float* __restrict__ aw, float* __restrict__ xh, float* __restrict__ sc)
{
    const int s = blockIdx.x, t = threadIdx.x;
    __shared__ float red[256];
    float lsum = 0.f;
#pragma unroll
    for (int i = 0; i < HID / 256; ++i) {
        int h = t + i * 256;
        float v = x[(size_t)s * HID + h] + hpe[(size_t)s * HID + h];
        xh[(size_t)s * HID + h] = v;
        lsum += v * aw[h];
    }
    red[t] = lsum; __syncthreads();
    for (int o = 128; o > 0; o >>= 1) { if (t < o) red[t] += red[t + o]; __syncthreads(); }
    if (t == 0) sc[s] = red[0];
}

__global__ __launch_bounds__(1024) void softmax_s_kernel(
    const float* __restrict__ sc, float* __restrict__ pr)
{
    __shared__ float red[1024];
    const int t = threadIdx.x;
    float v[SEQ / 1024];
    float lm = -1e30f;
#pragma unroll
    for (int i = 0; i < SEQ / 1024; ++i) { v[i] = sc[t + i * 1024]; lm = fmaxf(lm, v[i]); }
    red[t] = lm; __syncthreads();
    for (int o = 512; o > 0; o >>= 1) { if (t < o) red[t] = fmaxf(red[t], red[t + o]); __syncthreads(); }
    const float m = red[0]; __syncthreads();
    float ls = 0.f;
#pragma unroll
    for (int i = 0; i < SEQ / 1024; ++i) { v[i] = __expf(v[i] - m); ls += v[i]; }
    red[t] = ls; __syncthreads();
    for (int o = 512; o > 0; o >>= 1) { if (t < o) red[t] += red[t + o]; __syncthreads(); }
    const float inv = 1.f / red[0];
#pragma unroll
    for (int i = 0; i < SEQ / 1024; ++i) pr[t + i * 1024] = v[i] * inv;
}

__global__ __launch_bounds__(256) void pool_kernel(
    const float* __restrict__ xh, const float* __restrict__ pr,
    float* __restrict__ pooled)
{
    const int j = blockIdx.x, t = threadIdx.x;
    __shared__ float red[256];
    float ls = 0.f;
    for (int i = t; i < SEQ; i += 256) ls += xh[(size_t)i * HID + j] * pr[i];
    red[t] = ls; __syncthreads();
    for (int o = 128; o > 0; o >>= 1) { if (t < o) red[t] += red[t + o]; __syncthreads(); }
    if (t == 0) pooled[j] = red[0];
}

__global__ __launch_bounds__(256) void cls_kernel(
    const float* __restrict__ pooled, const float* __restrict__ Wc,
    const float* __restrict__ bc, float* __restrict__ out)
{
    const int c = blockIdx.x, t = threadIdx.x;
    __shared__ float red[256];
    float ls = 0.f;
#pragma unroll
    for (int i = 0; i < HID / 256; ++i) {
        int h = t + i * 256;
        ls += pooled[h] * Wc[(size_t)h * NCLS + c];
    }
    red[t] = ls; __syncthreads();
    for (int o = 128; o > 0; o >>= 1) { if (t < o) red[t] += red[t + o]; __syncthreads(); }
    if (t == 0) out[c] = red[0] + bc[c];
}

// ---------------- host orchestration ----------------
extern "C" void kernel_launch(void* const* d_in, const int* in_sizes, int n_in,
                              void* d_out, int out_size, void* d_ws, size_t ws_size,
                              hipStream_t stream)
{
    (void)in_sizes; (void)n_in; (void)out_size; (void)ws_size;
    const int*   ids   = (const int*)d_in[0];
    const int*   amask = (const int*)d_in[1];
    const float* we    = (const float*)d_in[2];
    const float* pe    = (const float*)d_in[3];
    const float* elng  = (const float*)d_in[4];
    const float* elnb  = (const float*)d_in[5];
    const float* Wq    = (const float*)d_in[6];
    const float* bq    = (const float*)d_in[7];
    const float* Wk    = (const float*)d_in[8];
    const float* bk    = (const float*)d_in[9];
    const float* Wv    = (const float*)d_in[10];
    const float* bv    = (const float*)d_in[11];
    const float* Wo    = (const float*)d_in[12];
    const float* bo    = (const float*)d_in[13];
    const float* alg   = (const float*)d_in[14];
    const float* alb   = (const float*)d_in[15];
    const float* Wf1   = (const float*)d_in[16];
    const float* bf1   = (const float*)d_in[17];
    const float* Wf2   = (const float*)d_in[18];
    const float* bf2   = (const float*)d_in[19];
    const float* flg   = (const float*)d_in[20];
    const float* flb   = (const float*)d_in[21];
    const float* hpe   = (const float*)d_in[22];
    const float* aw    = (const float*)d_in[23];
    const float* Wc    = (const float*)d_in[24];
    const float* bc    = (const float*)d_in[25];
    float* out = (float*)d_out;

    const size_t SH = (size_t)SEQ * HID;
    const size_t HH = (size_t)HID * HID;
    const size_t HF = (size_t)HID * FFN;

    char* p = (char*)d_ws;
    auto alloc = [&](size_t bytes) -> void* {
        void* r = (void*)p;
        p += (bytes + 255) & ~(size_t)255;
        return r;
    };
    float*    x    = (float*)alloc(SH * 4);          // fp32 hidden state
    float*    tb   = (float*)alloc(SH * 4);          // fp32 proj temp / head xh
    float*    scs  = (float*)alloc(SEQ * 4);         // pooling scores
    float*    prb  = (float*)alloc(SEQ * 4);         // pooling probs
    float*    pld  = (float*)alloc(HID * 4);         // pooled vector
    _Float16* x16  = (_Float16*)alloc(SH * 2);       // fp16 hidden
    _Float16* q16  = (_Float16*)alloc(SH * 2);
    _Float16* k16  = (_Float16*)alloc(SH * 2);
    _Float16* vt16 = (_Float16*)alloc(SH * 2);       // V transposed [H][S]
    _Float16* ab16 = (_Float16*)alloc(SH * 2);       // attention output
    _Float16* f116 = (_Float16*)alloc((size_t)SEQ * FFN * 2);
    _Float16* wq16 = (_Float16*)alloc(NLAY * HH * 2);   // all pre-transposed [N][K]
    _Float16* wk16 = (_Float16*)alloc(NLAY * HH * 2);
    _Float16* wv16 = (_Float16*)alloc(NLAY * HH * 2);
    _Float16* wo16 = (_Float16*)alloc(NLAY * HH * 2);
    _Float16* wf1t = (_Float16*)alloc(NLAY * HF * 2);   // [FF][H] per layer
    _Float16* wf2t = (_Float16*)alloc(NLAY * HF * 2);   // [H][FF] per layer

    // one-time (per launch) weight conversion + transpose to fp16
    {
        const int gHH = (int)((HH + 255) / 256), gHF = (int)((HF + 255) / 256);
        for (int l = 0; l < NLAY; ++l) {
            cvt_transpose_kernel<<<gHH, 256, 0, stream>>>(Wq + l * HH, wq16 + l * HH, HID, HID);
            cvt_transpose_kernel<<<gHH, 256, 0, stream>>>(Wk + l * HH, wk16 + l * HH, HID, HID);
            cvt_transpose_kernel<<<gHH, 256, 0, stream>>>(Wv + l * HH, wv16 + l * HH, HID, HID);
            cvt_transpose_kernel<<<gHH, 256, 0, stream>>>(Wo + l * HH, wo16 + l * HH, HID, HID);
            cvt_transpose_kernel<<<gHF, 256, 0, stream>>>(Wf1 + l * HF, wf1t + l * HF, HID, FFN);
            cvt_transpose_kernel<<<gHF, 256, 0, stream>>>(Wf2 + l * HF, wf2t + l * HF, FFN, HID);
        }
    }

    embed_ln_kernel<<<SEQ, 256, 0, stream>>>(ids, we, pe, elng, elnb, x, x16);

    const dim3 gP(HID / 128, SEQ / 128);   // proj GEMMs (N = 768)
    const dim3 gF(FFN / 128, SEQ / 128);   // FFN1 GEMM  (N = 3072)
    for (int l = 0; l < NLAY; ++l) {
        gemm16_kernel<0, 1><<<gP, 256, 0, stream>>>(x16, wq16 + l * HH, bq + l * HID, q16, SEQ, HID, HID);
        gemm16_kernel<0, 1><<<gP, 256, 0, stream>>>(x16, wk16 + l * HH, bk + l * HID, k16, SEQ, HID, HID);
        gemm16_kernel<0, 2><<<gP, 256, 0, stream>>>(x16, wv16 + l * HH, bv + l * HID, vt16, SEQ, HID, HID);

        attn_kernel<<<dim3(NHEAD, NBLK), 256, 0, stream>>>(q16, k16, vt16, amask, ab16);

        gemm16_kernel<0, 0><<<gP, 256, 0, stream>>>(ab16, wo16 + l * HH, bo + l * HID, tb, SEQ, HID, HID);
        add_ln_kernel<<<SEQ, 256, 0, stream>>>(x, tb, alg + l * HID, alb + l * HID, x, x16);

        gemm16_kernel<1, 1><<<gF, 256, 0, stream>>>(x16, wf1t + l * HF, bf1 + l * FFN, f116, SEQ, HID, FFN);
        gemm16_kernel<0, 0><<<gP, 256, 0, stream>>>(f116, wf2t + l * HF, bf2 + l * HID, tb, SEQ, FFN, HID);
        add_ln_kernel<<<SEQ, 256, 0, stream>>>(x, tb, flg + l * HID, flb + l * HID, x, x16);
    }

    // head (xh reuses tb; q/k/v buffers are dead here)
    head_pos_score_kernel<<<SEQ, 256, 0, stream>>>(x, hpe, aw, tb, scs);
    softmax_s_kernel<<<1, 1024, 0, stream>>>(scs, prb);
    pool_kernel<<<HID, 256, 0, stream>>>(tb, prb, pld);
    cls_kernel<<<NCLS, 256, 0, stream>>>(pld, Wc, bc, out);
}
